// MultiSpeakerTacotron2_55061480734876
// MI455X (gfx1250) — matrix operand order repaired and checked by reading, not compiled, not run on hardware
//
#include <hip/hip_runtime.h>
#include <hip/hip_bf16.h>

typedef __bf16 bf16_t;
typedef __attribute__((ext_vector_type(16))) __bf16 v16bf;
typedef __attribute__((ext_vector_type(8)))  __bf16 v8bf;
typedef __attribute__((ext_vector_type(8)))  float  v8f;

#define B_SZ 64
#define S_LEN 400
#define ROWS (B_SZ * S_LEN)   // 25600 positions

// ---------------------------------------------------------------------------
// WMMA fragment helpers (CDNA5 ISA 7.12.2 layouts, wave32)
// A 16x32 bf16: lane L -> row M=L&15, half=L>>4; elems j<8 : K=half*8+j
//                                              elems j>=8: K=16+half*8+(j-8)
// B 32x16 bf16 from row-major W[N][K]: lane L -> col N=L&15, half=L>>4;
//                                      elems j: same K as A (table order)
// C/D f32: lane L -> N=L&15; vgpr r -> M = r + 8*(L>>4)
// ---------------------------------------------------------------------------
__device__ __forceinline__ v16bf frag_load_a(const bf16_t* p) {
  // p = &A[row][k0 + half*8]
  v8bf lo = *(const v8bf*)(p);
  v8bf hi = *(const v8bf*)(p + 16);
  v16bf r;
#pragma unroll
  for (int j = 0; j < 8; ++j) { r[j] = lo[j]; r[j + 8] = hi[j]; }
  return r;
}

__device__ __forceinline__ v16bf frag_load_b(const bf16_t* p) {
  // p = &W[n][k0 + half*16]; loads K = half*8+j and 16+half*8+j
  const bf16_t* q = p - ((threadIdx.x >> 4) & 1) * 8;
  v8bf lo = *(const v8bf*)(q);
  v8bf hi = *(const v8bf*)(q + 16);
  v16bf r;
#pragma unroll
  for (int j = 0; j < 8; ++j) { r[j] = lo[j]; r[j + 8] = hi[j]; }
  return r;
}

__device__ __forceinline__ v16bf frag_zero() {
  v16bf r;
#pragma unroll
  for (int j = 0; j < 16; ++j) r[j] = (bf16_t)0.0f;
  return r;
}

__device__ __forceinline__ v8f wmma_bf16(v16bf a, v16bf b, v8f c) {
  return __builtin_amdgcn_wmma_f32_16x16x32_bf16(false, a, false, b,
                                                 (short)0, c, false, false);
}

__device__ __forceinline__ float sigmoidf_(float x) {
  return 1.0f / (1.0f + __expf(-x));
}

// ---------------------------------------------------------------------------
// Small utility kernels
// ---------------------------------------------------------------------------
__global__ void tac2_repack(const float* __restrict__ src, bf16_t* __restrict__ dst,
                            int O, int I, int Ipad, int sStrideO, int sStrideI,
                            int sOffset) {
  int idx = blockIdx.x * blockDim.x + threadIdx.x;
  if (idx >= O * Ipad) return;
  int o = idx / Ipad, i = idx - o * Ipad;
  float v = (i < I) ? src[(size_t)sOffset + (size_t)o * sStrideO + (size_t)i * sStrideI]
                    : 0.0f;
  dst[idx] = (bf16_t)v;
}

__global__ void tac2_embed(const int* __restrict__ ids, const float* __restrict__ emb,
                           bf16_t* __restrict__ x0, int total) {
  int i = blockIdx.x * blockDim.x + threadIdx.x;
  if (i >= total) return;
  int r = i >> 9, c = i & 511;
  x0[i] = (bf16_t)emb[(size_t)ids[r] * 512 + c];
}

__global__ void tac2_zero_bf16(bf16_t* __restrict__ p, int n) {
  int i = blockIdx.x * blockDim.x + threadIdx.x;
  if (i < n) p[i] = (bf16_t)0.0f;
}

// ---------------------------------------------------------------------------
// Generic WMMA GEMM / 1D-conv kernel.
//   C[M,N](f32/bf16) = act( scale[n] * (sum_kw A[row+kw-pad, :K] @ W[kw][N][K]^T) + bias[n] )
// A row-major [M,K] bf16, W k-major slabs of row-major [N][K] bf16.
// seqLen>0 enables conv boundary masking (rows are b*seqLen+s).
// remapS>0 remaps output row r=b*S+s -> s*remapB+b (time-major output).
// Block = 256 thr = 8 waves (2x4), block tile 64x128, wave tile 32x32.
// The 32-row wave tile is contiguous, so conv-edge validity is decided
// wave-uniformly per kw: interior tiles run a clean unmasked K loop (no exec
// mask churn in the hot path), only boundary tiles take the masked loop.
// ---------------------------------------------------------------------------
__global__ __launch_bounds__(256)
void tac2_gemm(const bf16_t* __restrict__ A, const bf16_t* __restrict__ W,
               float* __restrict__ Cf, bf16_t* __restrict__ Cbf,
               const float* __restrict__ scale, const float* __restrict__ bias,
               int M, int N, int K, int KW, int pad, int seqLen,
               int act, int remapS, int remapB, int ldc) {
  const int lane = threadIdx.x & 31;
  const int wid  = threadIdx.x >> 5;
  const int lr   = lane & 15;
  const int half = lane >> 4;
  const int m0 = blockIdx.y * 64 + (wid >> 2) * 32;
  const int n0 = blockIdx.x * 128 + (wid & 3) * 32;

  v8f acc[2][2];
#pragma unroll
  for (int i = 0; i < 2; ++i)
#pragma unroll
    for (int j = 0; j < 2; ++j)
#pragma unroll
      for (int r = 0; r < 8; ++r) acc[i][j][r] = 0.0f;

  int rowA[2], sIn[2], nIdx[2];
#pragma unroll
  for (int i = 0; i < 2; ++i) {
    rowA[i] = m0 + i * 16 + lr;
    sIn[i] = (seqLen > 0) ? (rowA[i] % seqLen) : 0;
  }
#pragma unroll
  for (int j = 0; j < 2; ++j) {
    int n = n0 + j * 16 + lr;
    nIdx[j] = (n < N) ? n : (N - 1);
  }
  // wave-uniform: first row's position within its sequence; the 32-row tile
  // stays inside one sequence iff sLo+31 < seqLen (seqLen=400 >> 32).
  const int sLo = (seqLen > 0) ? (m0 % seqLen) : 0;

  for (int kw = 0; kw < KW; ++kw) {
    const int d = kw - pad;
    const bf16_t* Wk = W + (size_t)kw * N * K;
    const bf16_t* aBase0 = A + (size_t)(rowA[0] + d) * K + half * 8;
    const bf16_t* aBase1 = A + (size_t)(rowA[1] + d) * K + half * 8;
    const bf16_t* wBase0 = Wk + (size_t)nIdx[0] * K + half * 16;
    const bf16_t* wBase1 = Wk + (size_t)nIdx[1] * K + half * 16;

    const bool allValid =
        (seqLen <= 0) ||
        ((sLo + 31 < seqLen) && (sLo + d >= 0) && (sLo + 31 + d < seqLen));

    if (allValid) {
      // -------- fast path: no masking in the K loop --------
      for (int k0 = 0; k0 < K; k0 += 32) {
        __builtin_prefetch(wBase0 + k0 + K, 0, 1);
        v16bf a0 = frag_load_a(aBase0 + k0);
        v16bf a1 = frag_load_a(aBase1 + k0);
        v16bf b0 = frag_load_b(wBase0 + k0);
        v16bf b1 = frag_load_b(wBase1 + k0);
        acc[0][0] = wmma_bf16(a0, b0, acc[0][0]);
        acc[1][0] = wmma_bf16(a1, b0, acc[1][0]);
        acc[0][1] = wmma_bf16(a0, b1, acc[0][1]);
        acc[1][1] = wmma_bf16(a1, b1, acc[1][1]);
      }
    } else {
      // -------- boundary path: per-lane masking --------
      bool valid[2];
#pragma unroll
      for (int i = 0; i < 2; ++i)
        valid[i] = (seqLen <= 0) || ((unsigned)(sIn[i] + d) < (unsigned)seqLen);
      for (int k0 = 0; k0 < K; k0 += 32) {
        v16bf a0 = valid[0] ? frag_load_a(aBase0 + k0) : frag_zero();
        v16bf a1 = valid[1] ? frag_load_a(aBase1 + k0) : frag_zero();
        v16bf b0 = frag_load_b(wBase0 + k0);
        v16bf b1 = frag_load_b(wBase1 + k0);
        acc[0][0] = wmma_bf16(a0, b0, acc[0][0]);
        acc[1][0] = wmma_bf16(a1, b0, acc[1][0]);
        acc[0][1] = wmma_bf16(a0, b1, acc[0][1]);
        acc[1][1] = wmma_bf16(a1, b1, acc[1][1]);
      }
    }
  }

#pragma unroll
  for (int i = 0; i < 2; ++i) {
#pragma unroll
    for (int j = 0; j < 2; ++j) {
      int n = n0 + j * 16 + lr;
      if (n >= N) continue;
      float sc = scale ? scale[n] : 1.0f;
      float bi = bias ? bias[n] : 0.0f;
#pragma unroll
      for (int r = 0; r < 8; ++r) {
        int m = m0 + i * 16 + r + 8 * half;
        float v = acc[i][j][r] * sc + bi;
        if (act == 1) v = fmaxf(v, 0.0f);
        else if (act == 2) v = tanhf(v);
        int orow = m;
        if (remapS > 0) { int b = m / remapS, s = m - b * remapS; orow = s * remapB + b; }
        if (Cf)  Cf[(size_t)orow * ldc + n] = v;
        if (Cbf) Cbf[(size_t)orow * ldc + n] = (bf16_t)v;
      }
    }
  }
}

// ---------------------------------------------------------------------------
// Persistent BiLSTM encoder: grid = {2 blocks: fwd, bwd} x 1024 threads.
// h (bf16) and c (f32) live in LDS; gates x-part precomputed time-major.
// ---------------------------------------------------------------------------
__global__ __launch_bounds__(1024)
void tac2_bilstm(const bf16_t* __restrict__ gxF, const bf16_t* __restrict__ gxB,
                 const bf16_t* __restrict__ WhhF, const bf16_t* __restrict__ WhhB,
                 bf16_t* __restrict__ encO, float* __restrict__ gbuf) {
  const int dir = blockIdx.x;
  const bf16_t* gx  = dir ? gxB : gxF;
  const bf16_t* Whh = dir ? WhhB : WhhF;          // [1024][256] row-major
  float* gb = gbuf + (size_t)dir * B_SZ * 1024;

  extern __shared__ char smem[];
  bf16_t* hS = (bf16_t*)smem;                     // [64][256]
  float*  cS = (float*)(smem + B_SZ * 256 * sizeof(bf16_t));

  for (int idx = threadIdx.x; idx < B_SZ * 256; idx += 1024) {
    hS[idx] = (bf16_t)0.0f; cS[idx] = 0.0f;
  }
  __syncthreads();

  const int lane = threadIdx.x & 31;
  const int wid  = threadIdx.x >> 5;
  const int lr = lane & 15, half = lane >> 4;

  for (int step = 0; step < S_LEN; ++step) {
    const int t = dir ? (S_LEN - 1 - step) : step;
    // gates = gx[t] + h @ Whh^T  -> [64 x 1024], 256 tiles over 32 waves
    for (int tl = wid; tl < 256; tl += 32) {
      const int m0 = (tl >> 6) * 16;
      const int n0 = (tl & 63) * 16;
      v8f acc;
#pragma unroll
      for (int r = 0; r < 8; ++r) {
        int m = m0 + r + 8 * half, n = n0 + lr;
        acc[r] = (float)gx[((size_t)(t * B_SZ + m)) * 1024 + n];
      }
#pragma unroll
      for (int k0 = 0; k0 < 256; k0 += 32) {
        v16bf af = frag_load_a(hS + (m0 + lr) * 256 + k0 + half * 8);
        v16bf bf = frag_load_b(Whh + (size_t)(n0 + lr) * 256 + k0 + half * 16);
        acc = wmma_bf16(af, bf, acc);
      }
#pragma unroll
      for (int r = 0; r < 8; ++r) {
        int m = m0 + r + 8 * half, n = n0 + lr;
        gb[m * 1024 + n] = acc[r];
      }
    }
    __threadfence_block();
    __syncthreads();
    // LSTM cell elementwise
    for (int idx = threadIdx.x; idx < B_SZ * 256; idx += 1024) {
      int m = idx >> 8, n = idx & 255;
      float gi = gb[m * 1024 + n];
      float gf = gb[m * 1024 + 256 + n];
      float gg = gb[m * 1024 + 512 + n];
      float go = gb[m * 1024 + 768 + n];
      float c2 = sigmoidf_(gf) * cS[idx] + sigmoidf_(gi) * tanhf(gg);
      float hv = sigmoidf_(go) * tanhf(c2);
      cS[idx] = c2;
      hS[idx] = (bf16_t)hv;
      encO[((size_t)(t * B_SZ + m)) * 512 + dir * 256 + n] = (bf16_t)hv;
    }
    __syncthreads();
  }
}

// ---------------------------------------------------------------------------
// Persistent autoregressive decoder: 1 block x 1024 threads.
// Both hidden states (64x1024 bf16) resident in 256KB LDS. Note the reference
// passes (h,h) as (h,c): the cell state each step IS the previous hidden.
// ---------------------------------------------------------------------------
__global__ __launch_bounds__(1024)
void tac2_decoder(const bf16_t* __restrict__ attgx, const float* __restrict__ spkg,
                  const bf16_t* __restrict__ attWhh,
                  const bf16_t* __restrict__ decWih, const bf16_t* __restrict__ decWhh,
                  const float* __restrict__ dec_b,
                  const bf16_t* __restrict__ projW, const float* __restrict__ proj_b,
                  float* __restrict__ gA, float* __restrict__ gD,
                  float* __restrict__ mel_out, bf16_t* __restrict__ pbuf) {
  extern __shared__ char smem[];
  bf16_t* hA = (bf16_t*)smem;            // [64][1024]
  bf16_t* hD = hA + B_SZ * 1024;         // [64][1024]

  for (int idx = threadIdx.x; idx < B_SZ * 1024; idx += 1024) {
    hA[idx] = (bf16_t)0.0f; hD[idx] = (bf16_t)0.0f;
  }
  __syncthreads();

  const int lane = threadIdx.x & 31;
  const int wid  = threadIdx.x >> 5;
  const int lr = lane & 15, half = lane >> 4;

  for (int t = 0; t < S_LEN; ++t) {
    // ---- attention-LSTM gates: [64 x 4096] = attgx[t] + spkg + hA @ attWhh^T
    for (int tl = wid; tl < 1024; tl += 32) {
      const int m0 = (tl >> 8) * 16;
      const int n0 = (tl & 255) * 16;
      v8f acc;
#pragma unroll
      for (int r = 0; r < 8; ++r) {
        int m = m0 + r + 8 * half, n = n0 + lr;
        acc[r] = (float)attgx[((size_t)(t * B_SZ + m)) * 4096 + n] + spkg[m * 4096 + n];
      }
      for (int k0 = 0; k0 < 1024; k0 += 32) {
        v16bf af = frag_load_a(hA + (m0 + lr) * 1024 + k0 + half * 8);
        v16bf bf = frag_load_b(attWhh + (size_t)(n0 + lr) * 1024 + k0 + half * 16);
        acc = wmma_bf16(af, bf, acc);
      }
#pragma unroll
      for (int r = 0; r < 8; ++r) {
        int m = m0 + r + 8 * half, n = n0 + lr;
        gA[m * 4096 + n] = acc[r];
      }
    }
    __threadfence_block();
    __syncthreads();
    for (int idx = threadIdx.x; idx < B_SZ * 1024; idx += 1024) {
      int m = idx >> 10, n = idx & 1023;
      float gi = gA[m * 4096 + n],        gf = gA[m * 4096 + 1024 + n];
      float gg = gA[m * 4096 + 2048 + n], go = gA[m * 4096 + 3072 + n];
      float c2 = sigmoidf_(gf) * (float)hA[idx] + sigmoidf_(gi) * tanhf(gg);
      hA[idx] = (bf16_t)(sigmoidf_(go) * tanhf(c2));
    }
    __syncthreads();
    // ---- decoder-LSTM gates: dec_b + hA @ decWih^T + hD @ decWhh^T
    for (int tl = wid; tl < 1024; tl += 32) {
      const int m0 = (tl >> 8) * 16;
      const int n0 = (tl & 255) * 16;
      const int n  = n0 + lr;
      v8f acc;
#pragma unroll
      for (int r = 0; r < 8; ++r) acc[r] = dec_b[n];
      for (int k0 = 0; k0 < 1024; k0 += 32) {
        v16bf a1 = frag_load_a(hA + (m0 + lr) * 1024 + k0 + half * 8);
        v16bf b1 = frag_load_b(decWih + (size_t)n * 1024 + k0 + half * 16);
        acc = wmma_bf16(a1, b1, acc);
        v16bf a2 = frag_load_a(hD + (m0 + lr) * 1024 + k0 + half * 8);
        v16bf b2 = frag_load_b(decWhh + (size_t)n * 1024 + k0 + half * 16);
        acc = wmma_bf16(a2, b2, acc);
      }
#pragma unroll
      for (int r = 0; r < 8; ++r) {
        int m = m0 + r + 8 * half;
        gD[m * 4096 + n] = acc[r];
      }
    }
    __threadfence_block();
    __syncthreads();
    for (int idx = threadIdx.x; idx < B_SZ * 1024; idx += 1024) {
      int m = idx >> 10, n = idx & 1023;
      float gi = gD[m * 4096 + n],        gf = gD[m * 4096 + 1024 + n];
      float gg = gD[m * 4096 + 2048 + n], go = gD[m * 4096 + 3072 + n];
      float c2 = sigmoidf_(gf) * (float)hD[idx] + sigmoidf_(gi) * tanhf(gg);
      hD[idx] = (bf16_t)(sigmoidf_(go) * tanhf(c2));
    }
    __syncthreads();
    // ---- mel projection: [64 x 80] = hD @ projW^T + proj_b (20 tiles)
    if (wid < 20) {
      const int m0 = (wid & 3) * 16;
      const int n0 = (wid >> 2) * 16;
      const int n  = n0 + lr;                       // n < 80 always
      v8f acc;
#pragma unroll
      for (int r = 0; r < 8; ++r) acc[r] = proj_b[n];
      for (int k0 = 0; k0 < 1024; k0 += 32) {
        v16bf af = frag_load_a(hD + (m0 + lr) * 1024 + k0 + half * 8);
        v16bf bf = frag_load_b(projW + (size_t)n * 1024 + k0 + half * 16);
        acc = wmma_bf16(af, bf, acc);
      }
#pragma unroll
      for (int r = 0; r < 8; ++r) {
        int m = m0 + r + 8 * half;
        float v = acc[r];
        mel_out[((size_t)m * S_LEN + t) * 80 + n] = v;              // [B,S,80]
        pbuf[((size_t)m * S_LEN + t) * 96 + n]   = (bf16_t)v;       // padded bf16
      }
    }
    __syncthreads();
  }
}

// ---------------------------------------------------------------------------
// Host orchestration
// ---------------------------------------------------------------------------
extern "C" void kernel_launch(void* const* d_in, const int* in_sizes, int n_in,
                              void* d_out, int out_size, void* d_ws, size_t ws_size,
                              hipStream_t stream) {
  (void)in_sizes; (void)n_in; (void)out_size; (void)ws_size;

  const int*   text_ids = (const int*)  d_in[0];
  const float* spk      = (const float*)d_in[1];
  const float* emb      = (const float*)d_in[2];
  const float* enc_w    = (const float*)d_in[3];
  const float* enc_g    = (const float*)d_in[4];
  const float* enc_b    = (const float*)d_in[5];
  const float* lfWih    = (const float*)d_in[6];
  const float* lfWhh    = (const float*)d_in[7];
  const float* lfB      = (const float*)d_in[8];
  const float* lbWih    = (const float*)d_in[9];
  const float* lbWhh    = (const float*)d_in[10];
  const float* lbB      = (const float*)d_in[11];
  const float* aWih     = (const float*)d_in[12];
  const float* aWhh_f   = (const float*)d_in[13];
  const float* aB       = (const float*)d_in[14];
  const float* dWih_f   = (const float*)d_in[15];
  const float* dWhh_f   = (const float*)d_in[16];
  const float* dB       = (const float*)d_in[17];
  const float* pWf      = (const float*)d_in[18];
  const float* pB       = (const float*)d_in[19];
  const float* pn_w0    = (const float*)d_in[20];
  const float* pn_g0    = (const float*)d_in[21];
  const float* pn_b0    = (const float*)d_in[22];
  const float* pn_wm    = (const float*)d_in[23];
  const float* pn_gm    = (const float*)d_in[24];
  const float* pn_bm    = (const float*)d_in[25];
  const float* pn_w4    = (const float*)d_in[26];
  const float* pn_g4    = (const float*)d_in[27];
  const float* pn_b4    = (const float*)d_in[28];
  float* out = (float*)d_out;

  // workspace carve-up
  char* ws = (char*)d_ws;
  size_t off = 0;
  auto alloc = [&](size_t bytes) -> char* {
    char* p = ws + off;
    off = (off + bytes + 255) & ~(size_t)255;
    return p;
  };
  bf16_t* x0    = (bf16_t*)alloc((size_t)ROWS * 512 * 2);
  bf16_t* actA  = (bf16_t*)alloc((size_t)ROWS * 512 * 2);
  bf16_t* actB  = (bf16_t*)alloc((size_t)ROWS * 512 * 2);
  bf16_t* gxf   = (bf16_t*)alloc((size_t)ROWS * 1024 * 2);
  bf16_t* gxb   = (bf16_t*)alloc((size_t)ROWS * 1024 * 2);
  bf16_t* encO  = (bf16_t*)alloc((size_t)ROWS * 512 * 2);
  bf16_t* agx   = (bf16_t*)alloc((size_t)ROWS * 4096 * 2);
  float*  spkg  = (float*) alloc((size_t)B_SZ * 4096 * 4);
  float*  gbuf  = (float*) alloc((size_t)2 * B_SZ * 1024 * 4);
  float*  gA    = (float*) alloc((size_t)B_SZ * 4096 * 4);
  float*  gD    = (float*) alloc((size_t)B_SZ * 4096 * 4);
  bf16_t* pbuf  = (bf16_t*)alloc((size_t)ROWS * 96 * 2);
  bf16_t* pactA = (bf16_t*)alloc((size_t)ROWS * 512 * 2);
  bf16_t* pactB = (bf16_t*)alloc((size_t)ROWS * 512 * 2);
  // repacked bf16 weights, row-major [N][K] (k-major slabs for convs)
  bf16_t* encW  = (bf16_t*)alloc((size_t)3 * 5 * 512 * 512 * 2);
  bf16_t* WihF  = (bf16_t*)alloc((size_t)1024 * 512 * 2);
  bf16_t* WihB  = (bf16_t*)alloc((size_t)1024 * 512 * 2);
  bf16_t* WhhF  = (bf16_t*)alloc((size_t)1024 * 256 * 2);
  bf16_t* WhhB  = (bf16_t*)alloc((size_t)1024 * 256 * 2);
  bf16_t* aWihE = (bf16_t*)alloc((size_t)4096 * 512 * 2);
  bf16_t* aWihS = (bf16_t*)alloc((size_t)4096 * 256 * 2);
  bf16_t* aWhh  = (bf16_t*)alloc((size_t)4096 * 1024 * 2);
  bf16_t* dWih  = (bf16_t*)alloc((size_t)4096 * 1024 * 2);
  bf16_t* dWhh  = (bf16_t*)alloc((size_t)4096 * 1024 * 2);
  bf16_t* projW = (bf16_t*)alloc((size_t)80 * 1024 * 2);
  bf16_t* pnW0  = (bf16_t*)alloc((size_t)5 * 512 * 96 * 2);
  bf16_t* pnWM  = (bf16_t*)alloc((size_t)3 * 5 * 512 * 512 * 2);
  bf16_t* pnW4  = (bf16_t*)alloc((size_t)5 * 80 * 512 * 2);
  bf16_t* spkbf = (bf16_t*)alloc((size_t)B_SZ * 256 * 2);

  auto rp = [&](const float* s, bf16_t* d, int O, int I, int Ipad,
                int so, int si, int ofs) {
    int total = O * Ipad;
    tac2_repack<<<(total + 255) / 256, 256, 0, stream>>>(s, d, O, I, Ipad, so, si, ofs);
  };
  // encoder conv weights [3][512][512][5] -> [l][kw][512][512]
  for (int l = 0; l < 3; ++l)
    for (int k = 0; k < 5; ++k)
      rp(enc_w, encW + ((size_t)(l * 5 + k)) * 512 * 512, 512, 512, 512,
         512 * 5, 5, l * 512 * 512 * 5 + k);
  rp(lfWih, WihF, 1024, 512, 512, 512, 1, 0);
  rp(lbWih, WihB, 1024, 512, 512, 512, 1, 0);
  rp(lfWhh, WhhF, 1024, 256, 256, 256, 1, 0);
  rp(lbWhh, WhhB, 1024, 256, 256, 256, 1, 0);
  rp(aWih,  aWihE, 4096, 512, 512, 768, 1, 0);     // enc_out part of att input
  rp(aWih,  aWihS, 4096, 256, 256, 768, 1, 512);   // speaker part of att input
  rp(aWhh_f, aWhh, 4096, 1024, 1024, 1024, 1, 0);
  rp(dWih_f, dWih, 4096, 1024, 1024, 1024, 1, 0);
  rp(dWhh_f, dWhh, 4096, 1024, 1024, 1024, 1, 0);
  rp(pWf,    projW, 80, 1024, 1024, 1024, 1, 0);
  for (int k = 0; k < 5; ++k)
    rp(pn_w0, pnW0 + (size_t)k * 512 * 96, 512, 80, 96, 80 * 5, 5, k);
  for (int l = 0; l < 3; ++l)
    for (int k = 0; k < 5; ++k)
      rp(pn_wm, pnWM + ((size_t)(l * 5 + k)) * 512 * 512, 512, 512, 512,
         512 * 5, 5, l * 512 * 512 * 5 + k);
  for (int k = 0; k < 5; ++k)
    rp(pn_w4, pnW4 + (size_t)k * 80 * 512, 80, 512, 512, 512 * 5, 5, k);
  rp(spk, spkbf, B_SZ, 256, 256, 256, 1, 0);

  tac2_embed<<<(ROWS * 512 + 255) / 256, 256, 0, stream>>>(text_ids, emb, x0, ROWS * 512);
  tac2_zero_bf16<<<(ROWS * 96 + 255) / 256, 256, 0, stream>>>(pbuf, ROWS * 96);

  auto gemm = [&](const bf16_t* A, const bf16_t* W, float* Cf, bf16_t* Cbf,
                  const float* scale, const float* bias, int M, int N, int K,
                  int KW, int pad, int seqLen, int act, int remapS, int remapB,
                  int ldc) {
    dim3 g((N + 127) / 128, (M + 63) / 64);
    tac2_gemm<<<g, 256, 0, stream>>>(A, W, Cf, Cbf, scale, bias, M, N, K,
                                     KW, pad, seqLen, act, remapS, remapB, ldc);
  };

  // encoder: 3x  relu(conv * gamma + beta)
  gemm(x0,   encW,                          nullptr, actA, enc_g,        enc_b,
       ROWS, 512, 512, 5, 2, S_LEN, 1, 0, 0, 512);
  gemm(actA, encW + (size_t)5 * 512 * 512,  nullptr, actB, enc_g + 512,  enc_b + 512,
       ROWS, 512, 512, 5, 2, S_LEN, 1, 0, 0, 512);
  gemm(actB, encW + (size_t)10 * 512 * 512, nullptr, actA, enc_g + 1024, enc_b + 1024,
       ROWS, 512, 512, 5, 2, S_LEN, 1, 0, 0, 512);

  // BiLSTM x-projections (time-major output), then the persistent recurrence
  gemm(actA, WihF, nullptr, gxf, nullptr, lfB, ROWS, 1024, 512, 1, 0, 0, 0, S_LEN, B_SZ, 1024);
  gemm(actA, WihB, nullptr, gxb, nullptr, lbB, ROWS, 1024, 512, 1, 0, 0, 0, S_LEN, B_SZ, 1024);
  (void)hipFuncSetAttribute(reinterpret_cast<const void*>(tac2_bilstm),
                            hipFuncAttributeMaxDynamicSharedMemorySize, 98304);
  tac2_bilstm<<<2, 1024, B_SZ * 256 * 2 + B_SZ * 256 * 4, stream>>>(
      gxf, gxb, WhhF, WhhB, encO, gbuf);

  // decoder x-projections (hoisted out of the serial loop)
  gemm(encO,  aWihE, nullptr, agx, nullptr, aB, ROWS, 4096, 512, 1, 0, 0, 0, 0, 0, 4096);
  gemm(spkbf, aWihS, spkg, nullptr, nullptr, nullptr, B_SZ, 4096, 256, 1, 0, 0, 0, 0, 0, 4096);

  (void)hipFuncSetAttribute(reinterpret_cast<const void*>(tac2_decoder),
                            hipFuncAttributeMaxDynamicSharedMemorySize, 262144);
  tac2_decoder<<<1, 1024, 2 * B_SZ * 1024 * 2, stream>>>(
      agx, spkg, aWhh, dWih, dWhh, dB, projW, pB, gA, gD, out, pbuf);

  // postnet: tanh convs, last layer linear, written to second output half
  gemm(pbuf,  pnW0,                          nullptr, pactA, pn_g0,        pn_b0,
       ROWS, 512, 96, 5, 2, S_LEN, 2, 0, 0, 512);
  gemm(pactA, pnWM,                          nullptr, pactB, pn_gm,        pn_bm,
       ROWS, 512, 512, 5, 2, S_LEN, 2, 0, 0, 512);
  gemm(pactB, pnWM + (size_t)5 * 512 * 512,  nullptr, pactA, pn_gm + 512,  pn_bm + 512,
       ROWS, 512, 512, 5, 2, S_LEN, 2, 0, 0, 512);
  gemm(pactA, pnWM + (size_t)10 * 512 * 512, nullptr, pactB, pn_gm + 1024, pn_bm + 1024,
       ROWS, 512, 512, 5, 2, S_LEN, 2, 0, 0, 512);
  gemm(pactB, pnW4, out + (size_t)ROWS * 80, nullptr, pn_g4, pn_b4,
       ROWS, 80, 512, 5, 2, S_LEN, 0, 0, 0, 80);
}
